// Mpu2TextClassifier_10548439679613
// MI455X (gfx1250) — matrix operand order repaired and checked by reading, not compiled, not run on hardware
//
#include <hip/hip_runtime.h>

typedef _Float16 f16;
typedef _Float16 v16h __attribute__((ext_vector_type(16)));
typedef _Float16 v8h  __attribute__((ext_vector_type(8)));
typedef float    v8f  __attribute__((ext_vector_type(8)));

#define F_ACCUM 1
#define F_RELU  2
#define F_WC    4
#define F_WC16  8

// ---------------------------------------------------------------- fragments
// A (16x32 f16, row-major source): lane holds row m0+(lane&15).
// element e -> k = kh + (e<8 ? e : e+8), kh = (lane>=16)*8.
// -> two 16B loads at p and p+16, p = rowptr + k0 + kh.
__device__ __forceinline__ v16h load_a_frag(const f16* p) {
  v8h lo = *(const v8h*)p;
  v8h hi = *(const v8h*)(p + 16);
  v16h f;
#pragma unroll
  for (int i = 0; i < 8; ++i) { f[i] = lo[i]; f[i + 8] = hi[i]; }
  return f;
}

// B (32x16 f16) from Bt[N,K] row-major: lane holds col n0+(lane&15).
// element e -> k = kb + (e<8 ? e : e+8), kb = (lane>=16)*8.
__device__ __forceinline__ v16h load_b_frag(const f16* q) {
  v8h lo = *(const v8h*)q;
  v8h hi = *(const v8h*)(q + 16);
  v16h f;
#pragma unroll
  for (int i = 0; i < 8; ++i) { f[i] = lo[i]; f[i + 8] = hi[i]; }
  return f;
}

// ---------------------------------------------------------------- WMMA GEMM
// C[M,N] = sum_tap Amap_tap[M,K](f16) x Bt_tap[N,K]^T(f16)
// A row mapping: row m -> b = m/rob, r = m%rob, src = r*astride + ashift0 + tap;
// invalid src rows read from the zero row (zrow, >= K f16 of zeros).
// Bt per-tap plane stride = N*K (conv weights laid out [tap][N][K]).
// Workgroup = 8 waves = 64x128 C tile; each wave = 16x64 (4 accumulators).
__global__ __launch_bounds__(256)
void gemm_wmma_f16(const f16* __restrict__ A, const f16* __restrict__ Bt,
                   const f16* __restrict__ zrow,
                   float* C, f16* C16,
                   const float* __restrict__ bias, const float* residual,
                   int M, int N, int K,
                   int rob, int rib, int astride, int ashift0,
                   int ntaps, int flags)
{
  const int lane = threadIdx.x & 31;
  const int wid  = threadIdx.x >> 5;
  const int m0 = blockIdx.y * 64 + (wid & 3) * 16;
  const int n0 = blockIdx.x * 128 + (wid >> 2) * 64;
  if (m0 >= M || n0 >= N) return;           // wave-uniform

  const int mrow = m0 + (lane & 15);
  const int bb = mrow / rob;
  const int r  = mrow - bb * rob;
  const int kh = (lane >> 4) << 3;
  const size_t sK = (size_t)K;

  const f16* bcol = Bt + (size_t)(n0 + (lane & 15)) * sK + ((lane >> 4) << 3);

  v8f acc0 = {}; v8f acc1 = {}; v8f acc2 = {}; v8f acc3 = {};

  for (int tap = 0; tap < ntaps; ++tap) {
    int sr = r * astride + ashift0 + tap;
    const f16* ap = (sr >= 0 && sr < rib)
        ? A + (size_t)(bb * rib + sr) * sK + kh
        : zrow + kh;                         // unconditional loads, no EXEC games
    const f16* bp = bcol + (size_t)tap * (size_t)N * sK;
    for (int k0 = 0; k0 < K; k0 += 32) {
      v16h af = load_a_frag(ap + k0);
      v16h b0 = load_b_frag(bp + k0);
      v16h b1 = load_b_frag(bp + 16 * sK + k0);
      v16h b2 = load_b_frag(bp + 32 * sK + k0);
      v16h b3 = load_b_frag(bp + 48 * sK + k0);
      acc0 = __builtin_amdgcn_wmma_f32_16x16x32_f16(false, af, false, b0, (short)0, acc0, false, false);
      acc1 = __builtin_amdgcn_wmma_f32_16x16x32_f16(false, af, false, b1, (short)0, acc1, false, false);
      acc2 = __builtin_amdgcn_wmma_f32_16x16x32_f16(false, af, false, b2, (short)0, acc2, false, false);
      acc3 = __builtin_amdgcn_wmma_f32_16x16x32_f16(false, af, false, b3, (short)0, acc3, false, false);
    }
  }

#pragma unroll
  for (int r8 = 0; r8 < 8; ++r8) {
    int m = m0 + r8 + ((lane >> 4) << 3);
    int n = n0 + (lane & 15);
#pragma unroll
    for (int cc = 0; cc < 4; ++cc) {
      int nn = n + cc * 16;
      float v = (cc == 0) ? acc0[r8] : (cc == 1) ? acc1[r8] : (cc == 2) ? acc2[r8] : acc3[r8];
      size_t idx = (size_t)m * (size_t)N + nn;
      if (flags & F_ACCUM) v += C[idx];
      if (bias)            v += bias[nn];
      if (flags & F_RELU)  v = fmaxf(v, 0.0f);
      if (residual)        v += residual[idx];
      if (flags & F_WC)    C[idx] = v;
      if (flags & F_WC16)  C16[idx] = (f16)v;
    }
  }
}

// ---------------------------------------------------------------- weight prep
__global__ __launch_bounds__(256)
void zero_f16_kernel(f16* __restrict__ p, int n)
{
  int i = blockIdx.x * 256 + threadIdx.x;
  if (i < n) p[i] = (f16)0.0f;
}

__global__ __launch_bounds__(256)
void transpose_f16_kernel(const float* __restrict__ src, f16* __restrict__ dst,
                          int K, int N)   // src [K,N] f32 -> dst [N,K] f16
{
  long long idx = (long long)blockIdx.x * 256 + threadIdx.x;
  long long total = (long long)K * N;
  if (idx >= total) return;
  int k = (int)(idx % K);
  long long n = idx / K;
  dst[idx] = (f16)src[(size_t)k * N + n];
}

__global__ __launch_bounds__(256)
void convw_f16_kernel(const float* __restrict__ src, f16* __restrict__ dst,
                      int Cout, int Cin, int KK)  // [Cout,Cin,KK] -> [KK][Cout][Cin]
{
  int idx = blockIdx.x * 256 + threadIdx.x;
  int total = KK * Cout * Cin;
  if (idx >= total) return;
  int plane = Cout * Cin;
  int kk = idx / plane;
  int rem = idx - kk * plane;
  int co = rem / Cin, ci = rem - co * Cin;
  dst[idx] = (f16)src[(size_t)(co * Cin + ci) * KK + kk];
}

// ---------------------------------------------------------------- pad mask
__global__ __launch_bounds__(256)
void padmask_kernel(const float* __restrict__ data, int* __restrict__ pm)
{
  int idx = blockIdx.x * 256 + threadIdx.x;   // over 64*120
  if (idx >= 64 * 120) return;
  int b = idx / 120, t = idx - b * 120;
  if (t == 0) { pm[idx] = 0; return; }
  int tt = t - 1;
  const float* x = data + (size_t)b * 6 * 1190;  // channel 0
  float w = 0.0f;
  for (int kk = 0; kk < 10; ++kk) {
    int j = tt * 10 + kk;
    float s2 = 0.0f;
    for (int d2 = -1; d2 <= 1; ++d2) {
      int i = j + d2; if (i < 0 || i >= 1190) continue;
      float s1 = 0.0f;
      for (int d1 = -1; d1 <= 1; ++d1) { int u = i + d1; if (u >= 0 && u < 1190) s1 += x[u]; }
      s2 += s1;
    }
    w += s2;
  }
  pm[idx] = (w == 0.0f) ? 1 : 0;
}

// ---------------------------------------------------------------- conv front-end
__global__ __launch_bounds__(256)
void conv1_kernel(const float* __restrict__ data, const float* __restrict__ w,
                  const float* __restrict__ bias, const float* __restrict__ g,
                  const float* __restrict__ be, const float* __restrict__ mean,
                  const float* __restrict__ var, f16* __restrict__ h1h)
{
  int idx = blockIdx.x * 256 + threadIdx.x;   // [B,L,C] channels-last
  if (idx >= 64 * 1190 * 512) return;
  int co = idx & 511;
  int bl = idx >> 9;
  int l = bl % 1190, b = bl / 1190;
  const float* dbase = data + (size_t)b * 6 * 1190;
  float acc = bias[co];
#pragma unroll
  for (int ci = 0; ci < 6; ++ci)
#pragma unroll
    for (int kk = 0; kk < 3; ++kk) {
      int u = l + kk - 1;
      if (u >= 0 && u < 1190)
        acc += dbase[ci * 1190 + u] * w[(co * 6 + ci) * 3 + kk];
    }
  float inv = g[co] * rsqrtf(var[co] + 1e-5f);
  float o = fmaxf(acc * inv + (be[co] - mean[co] * inv), 0.0f);
  h1h[idx] = (f16)o;
}

__global__ __launch_bounds__(256)
void conv2_post_kernel(const float* __restrict__ raw, const float* __restrict__ cb,
                       const float* __restrict__ g, const float* __restrict__ be,
                       const float* __restrict__ mean, const float* __restrict__ var,
                       const f16* __restrict__ h1h, f16* __restrict__ h2h)
{
  int idx = blockIdx.x * 256 + threadIdx.x;
  if (idx >= 64 * 1190 * 512) return;
  int co = idx & 511;
  float inv = g[co] * rsqrtf(var[co] + 1e-5f);
  float o = fmaxf((raw[idx] + cb[co]) * inv + (be[co] - mean[co] * inv), 0.0f) + (float)h1h[idx];
  h2h[idx] = (f16)o;
}

__global__ __launch_bounds__(256)
void build_x_kernel(const float* __restrict__ raw, const float* __restrict__ cb,
                    const float* __restrict__ g, const float* __restrict__ be,
                    const float* __restrict__ mean, const float* __restrict__ var,
                    const float* __restrict__ cls_token, const float* __restrict__ alpha,
                    float* __restrict__ x)
{
  int idx = blockIdx.x * 256 + threadIdx.x;   // over 64*120*512
  if (idx >= 7680 * 512) return;
  int c = idx & 511;
  int bt = idx >> 9;
  int t = bt % 120, b = bt / 120;
  float val;
  if (t == 0) val = cls_token[c];
  else {
    float inv = g[c] * rsqrtf(var[c] + 1e-5f);
    float r = raw[((size_t)b * 119 + (t - 1)) * 512 + c];
    val = (r + cb[c]) * inv + (be[c] - mean[c] * inv);
  }
  int i = c >> 1;
  float freq = powf(10000.0f, -2.0f * (float)i / 512.0f);
  float ang = (float)t * freq;
  float pe = (c & 1) ? cosf(ang) : sinf(ang);
  x[idx] = val + alpha[0] * pe;
}

// ---------------------------------------------------------------- layernorm
__global__ __launch_bounds__(256)
void layernorm_f16_kernel(const float* __restrict__ x, const float* __restrict__ g,
                          const float* __restrict__ b, f16* __restrict__ out)
{
  const int row = blockIdx.x;
  const float* xr = x + (size_t)row * 512;
  float s = 0.0f, ss = 0.0f;
  for (int c = threadIdx.x; c < 512; c += 256) { float v = xr[c]; s += v; ss += v * v; }
  for (int off = 16; off; off >>= 1) { s += __shfl_xor(s, off, 32); ss += __shfl_xor(ss, off, 32); }
  __shared__ float sred[8], ssred[8], stat[2];
  int wid = threadIdx.x >> 5, lane = threadIdx.x & 31;
  if (lane == 0) { sred[wid] = s; ssred[wid] = ss; }
  __syncthreads();
  if (threadIdx.x == 0) {
    float ts = 0.0f, tss = 0.0f;
    for (int i = 0; i < 8; ++i) { ts += sred[i]; tss += ssred[i]; }
    float mu = ts * (1.0f / 512.0f);
    float var = tss * (1.0f / 512.0f) - mu * mu;
    stat[0] = mu; stat[1] = rsqrtf(var + 1e-5f);
  }
  __syncthreads();
  float mu = stat[0], rs = stat[1];
  f16* orow = out + (size_t)row * 512;
  for (int c = threadIdx.x; c < 512; c += 256)
    orow[c] = (f16)((xr[c] - mu) * rs * g[c] + b[c]);
}

// ---------------------------------------------------------------- attention
// one workgroup per (batch, head); K,V staged in LDS; exact mask semantics.
__global__ __launch_bounds__(256)
void attention_kernel(const float* __restrict__ qkv, const int* __restrict__ padmask,
                      f16* __restrict__ o16)
{
  __shared__ float kb[120 * 64];
  __shared__ float vb[120 * 64];
  __shared__ float ab[8 * 120];
  const int b = blockIdx.x >> 3;
  const int h = blockIdx.x & 7;
  const float* base = qkv + (size_t)b * 120 * 1536 + h * 64;
  for (int i = threadIdx.x; i < 120 * 64; i += 256) {
    int t = i >> 6, d = i & 63;
    const float* p = base + (size_t)t * 1536 + d;
    kb[i] = p[512];
    vb[i] = p[1024];
  }
  __syncthreads();
  const int wid = threadIdx.x >> 5, lane = threadIdx.x & 31;
  const int* pm = padmask + b * 120;
  const float NEG = -__builtin_inff();
  for (int t = wid; t < 120; t += 8) {          // exactly 15 iters per wave
    const float* qrow = base + (size_t)t * 1536;
    int j3 = lane + 96;
    int j3c = j3 < 120 ? j3 : 0;
    float s0 = 0, s1 = 0, s2 = 0, s3 = 0;
    for (int e = 0; e < 64; ++e) {
      float qe = qrow[e];
      s0 += qe * kb[(lane      ) * 64 + e];
      s1 += qe * kb[(lane +  32) * 64 + e];
      s2 += qe * kb[(lane +  64) * 64 + e];
      s3 += qe * kb[(j3c       ) * 64 + e];
    }
    float sc[4] = { s0 * 0.125f, s1 * 0.125f, s2 * 0.125f, s3 * 0.125f };
#pragma unroll
    for (int jj = 0; jj < 4; ++jj) {
      int j = lane + jj * 32;
      if (j >= 120) { sc[jj] = NEG; continue; }
      int dt = t - j; if (dt < 0) dt = -dt;
      bool keep = (t == 0) || (j == 0) || (dt <= 3);   // local band + cls row/col
      if (!keep)      sc[jj] = NEG;       // local mask -> -inf (wins over pad)
      else if (pm[j]) sc[jj] = -100.0f;   // pad-but-not-local -> -100
    }
    float mx = fmaxf(fmaxf(sc[0], sc[1]), fmaxf(sc[2], sc[3]));
    for (int off = 16; off; off >>= 1) mx = fmaxf(mx, __shfl_xor(mx, off, 32));
    float esum = 0.0f, ev[4];
#pragma unroll
    for (int jj = 0; jj < 4; ++jj) {
      ev[jj] = (sc[jj] == NEG) ? 0.0f : expf(sc[jj] - mx);
      esum += ev[jj];
    }
    for (int off = 16; off; off >>= 1) esum += __shfl_xor(esum, off, 32);
    float inv = 1.0f / esum;
    float qz = pm[t] ? 0.0f : 1.0f;       // zero padded query rows
#pragma unroll
    for (int jj = 0; jj < 4; ++jj) {
      int j = lane + jj * 32;
      if (j < 120) ab[wid * 120 + j] = ev[jj] * inv * qz;
    }
    __syncthreads();
    for (int d = lane; d < 64; d += 32) {
      float acc = 0.0f;
      for (int j = 0; j < 120; ++j) acc += ab[wid * 120 + j] * vb[j * 64 + d];
      o16[((size_t)(b * 120 + t)) * 512 + h * 64 + d] = (f16)acc;
    }
    __syncthreads();
  }
}

// ---------------------------------------------------------------- classifier
__global__ __launch_bounds__(256)
void classifier_kernel(const float* __restrict__ x, const float* __restrict__ w,
                       const float* __restrict__ bias, float* __restrict__ out)
{
  int idx = blockIdx.x * 256 + threadIdx.x;   // over 7680*30
  if (idx >= 7680 * 30) return;
  int n = idx % 30, m = idx / 30;
  const float* xr = x + (size_t)m * 512;
  float acc = bias[n];
  for (int k = 0; k < 512; ++k) acc += xr[k] * w[k * 30 + n];
  out[idx] = acc;
}

// ---------------------------------------------------------------- host
extern "C" void kernel_launch(void* const* d_in, const int* in_sizes, int n_in,
                              void* d_out, int out_size, void* d_ws, size_t ws_size,
                              hipStream_t stream)
{
  (void)in_sizes; (void)n_in; (void)out_size; (void)ws_size;
  // depth-first dict-order of setup_inputs()
  const float* data    = (const float*)d_in[0];
  const float* c1_w    = (const float*)d_in[1];
  const float* c1_b    = (const float*)d_in[2];
  const float* bn1_g   = (const float*)d_in[3];
  const float* bn1_b   = (const float*)d_in[4];
  const float* bn1_m   = (const float*)d_in[5];
  const float* bn1_v   = (const float*)d_in[6];
  const float* c2_w    = (const float*)d_in[7];
  const float* c2_b    = (const float*)d_in[8];
  const float* bn2_g   = (const float*)d_in[9];
  const float* bn2_b   = (const float*)d_in[10];
  const float* bn2_m   = (const float*)d_in[11];
  const float* bn2_v   = (const float*)d_in[12];
  const float* c3_w    = (const float*)d_in[13];
  const float* c3_b    = (const float*)d_in[14];
  const float* bn3_g   = (const float*)d_in[15];
  const float* bn3_b   = (const float*)d_in[16];
  const float* bn3_m   = (const float*)d_in[17];
  const float* bn3_v   = (const float*)d_in[18];
  const float* cls_tok = (const float*)d_in[19];
  const float* alpha   = (const float*)d_in[20];
  const float* ln1_g   = (const float*)d_in[21];
  const float* ln1_b   = (const float*)d_in[22];
  const float* qkv_w   = (const float*)d_in[23];
  const float* qkv_b   = (const float*)d_in[24];
  const float* proj_w  = (const float*)d_in[25];
  const float* proj_b  = (const float*)d_in[26];
  const float* ln2_g   = (const float*)d_in[27];
  const float* ln2_b   = (const float*)d_in[28];
  const float* ff1_w   = (const float*)d_in[29];
  const float* ff1_b   = (const float*)d_in[30];
  const float* ff2_w   = (const float*)d_in[31];
  const float* ff2_b   = (const float*)d_in[32];
  const float* cls_w   = (const float*)d_in[33];
  const float* cls_b   = (const float*)d_in[34];

  // workspace layout (bytes)
  constexpr size_t OFF_H1H   = 0;                                  // 64*1190*512 f16
  constexpr size_t OFF_H2H   = OFF_H1H   + 77987840ull;            // 64*1190*512 f16
  constexpr size_t OFF_CRAW  = OFF_H2H   + 77987840ull;            // 64*1190*512 f32
  constexpr size_t OFF_X     = OFF_CRAW  + 155975680ull;           // 7680*512 f32
  constexpr size_t OFF_QKV   = OFF_X     + 15728640ull;            // 7680*1536 f32
  constexpr size_t OFF_S16A  = OFF_QKV   + 47185920ull;            // 7680*512 f16
  constexpr size_t OFF_S16B  = OFF_S16A  + 7864320ull;             // 7680*2048 f16
  constexpr size_t OFF_PM    = OFF_S16B  + 31457280ull;            // 7680 int
  constexpr size_t OFF_WQKV  = OFF_PM    + 30720ull;               // 8*1536*512 f16
  constexpr size_t OFF_WPROJ = OFF_WQKV  + 12582912ull;            // 8*512*512 f16
  constexpr size_t OFF_WFF1  = OFF_WPROJ + 4194304ull;             // 8*2048*512 f16
  constexpr size_t OFF_WFF2  = OFF_WFF1  + 16777216ull;            // 8*512*2048 f16
  constexpr size_t OFF_WC2   = OFF_WFF2  + 16777216ull;            // 3*512*512 f16
  constexpr size_t OFF_WC3   = OFF_WC2   + 1572864ull;             // 10*512*512 f16
  constexpr size_t OFF_ZROW  = OFF_WC3   + 5242880ull;             // 2048 f16 zeros

  char* ws = (char*)d_ws;
  f16*   h1h   = (f16*)(ws + OFF_H1H);
  f16*   h2h   = (f16*)(ws + OFF_H2H);
  float* craw  = (float*)(ws + OFF_CRAW);
  float* x     = (float*)(ws + OFF_X);
  float* qkvb  = (float*)(ws + OFF_QKV);
  f16*   s16a  = (f16*)(ws + OFF_S16A);
  f16*   s16b  = (f16*)(ws + OFF_S16B);
  int*   pm    = (int*)(ws + OFF_PM);
  f16*   wqkv  = (f16*)(ws + OFF_WQKV);
  f16*   wproj = (f16*)(ws + OFF_WPROJ);
  f16*   wff1  = (f16*)(ws + OFF_WFF1);
  f16*   wff2  = (f16*)(ws + OFF_WFF2);
  f16*   wc2   = (f16*)(ws + OFF_WC2);
  f16*   wc3   = (f16*)(ws + OFF_WC3);
  f16*   zrow  = (f16*)(ws + OFF_ZROW);

  const int MT = 7680;           // B*120
  const int NBL = 64 * 1190 * 512;

  // ---- prep: zero row, pad mask, weight conversion (f32 -> transposed f16)
  zero_f16_kernel<<<dim3((2048 + 255) / 256), 256, 0, stream>>>(zrow, 2048);
  padmask_kernel<<<dim3((7680 + 255) / 256), 256, 0, stream>>>(data, pm);
  convw_f16_kernel<<<dim3((3 * 512 * 512 + 255) / 256), 256, 0, stream>>>(c2_w, wc2, 512, 512, 3);
  convw_f16_kernel<<<dim3((10 * 512 * 512 + 255) / 256), 256, 0, stream>>>(c3_w, wc3, 512, 512, 10);
  for (int d = 0; d < 8; ++d) {
    transpose_f16_kernel<<<dim3((512 * 1536 + 255) / 256), 256, 0, stream>>>(
        qkv_w + (size_t)d * 512 * 1536, wqkv + (size_t)d * 1536 * 512, 512, 1536);
    transpose_f16_kernel<<<dim3((512 * 512 + 255) / 256), 256, 0, stream>>>(
        proj_w + (size_t)d * 512 * 512, wproj + (size_t)d * 512 * 512, 512, 512);
    transpose_f16_kernel<<<dim3((512 * 2048 + 255) / 256), 256, 0, stream>>>(
        ff1_w + (size_t)d * 512 * 2048, wff1 + (size_t)d * 2048 * 512, 512, 2048);
    transpose_f16_kernel<<<dim3((2048 * 512 + 255) / 256), 256, 0, stream>>>(
        ff2_w + (size_t)d * 2048 * 512, wff2 + (size_t)d * 512 * 2048, 2048, 512);
  }

  // ---- conv front-end
  conv1_kernel<<<dim3((NBL + 255) / 256), 256, 0, stream>>>(
      data, c1_w, c1_b, bn1_g, bn1_b, bn1_m, bn1_v, h1h);

  // conv2: single fused 3-tap GEMM  [B*1190,512] x 3x[512,512]
  gemm_wmma_f16<<<dim3(512 / 128, 76160 / 64), 256, 0, stream>>>(
      h1h, wc2, zrow, craw, nullptr, nullptr, nullptr,
      76160, 512, 512, 1190, 1190, 1, -1, 3, F_WC);
  conv2_post_kernel<<<dim3((NBL + 255) / 256), 256, 0, stream>>>(
      craw, c2_b, bn2_g, bn2_b, bn2_m, bn2_v, h1h, h2h);

  // conv3: single fused 10-tap strided GEMM  [B*119,512] x 10x[512,512]
  gemm_wmma_f16<<<dim3(512 / 128, 7616 / 64), 256, 0, stream>>>(
      h2h, wc3, zrow, craw, nullptr, nullptr, nullptr,
      7616, 512, 512, 119, 1190, 10, 0, 10, F_WC);
  build_x_kernel<<<dim3((MT * 512 + 255) / 256), 256, 0, stream>>>(
      craw, c3_b, bn3_g, bn3_b, bn3_m, bn3_v, cls_tok, alpha, x);

  // ---- transformer blocks
  for (int d = 0; d < 8; ++d) {
    layernorm_f16_kernel<<<dim3(MT), 256, 0, stream>>>(x, ln1_g + d * 512, ln1_b + d * 512, s16a);
    gemm_wmma_f16<<<dim3(1536 / 128, MT / 64), 256, 0, stream>>>(
        s16a, wqkv + (size_t)d * 1536 * 512, zrow, qkvb, nullptr,
        qkv_b + d * 1536, nullptr, MT, 1536, 512, MT, MT, 1, 0, 1, F_WC);
    attention_kernel<<<dim3(64 * 8), 256, 0, stream>>>(qkvb, pm, s16a);
    gemm_wmma_f16<<<dim3(512 / 128, MT / 64), 256, 0, stream>>>(
        s16a, wproj + (size_t)d * 512 * 512, zrow, x, nullptr,
        proj_b + d * 512, x, MT, 512, 512, MT, MT, 1, 0, 1, F_WC);   // residual add
    layernorm_f16_kernel<<<dim3(MT), 256, 0, stream>>>(x, ln2_g + d * 512, ln2_b + d * 512, s16a);
    gemm_wmma_f16<<<dim3(2048 / 128, MT / 64), 256, 0, stream>>>(
        s16a, wff1 + (size_t)d * 2048 * 512, zrow, nullptr, s16b,
        ff1_b + d * 2048, nullptr, MT, 2048, 512, MT, MT, 1, 0, 1, F_RELU | F_WC16);
    gemm_wmma_f16<<<dim3(512 / 128, MT / 64), 256, 0, stream>>>(
        s16b, wff2 + (size_t)d * 512 * 2048, zrow, x, nullptr,
        ff2_b + d * 512, x, MT, 512, 2048, MT, MT, 1, 0, 1, F_WC);   // residual add
  }

  classifier_kernel<<<dim3((MT * 30 + 255) / 256), 256, 0, stream>>>(
      x, cls_w, cls_b, (float*)d_out);
}
